// TemporalLayer_28475633172745
// MI455X (gfx1250) — compile-verified, matrix operand order checked by reading, not run on hardware
//
#include <hip/hip_runtime.h>

// ---------------------------------------------------------------------------
// Temporal causal self-attention layer for MI455X (gfx1250, wave32, WMMA).
// One workgroup (8 waves) per node; all per-node tiles staged in ~300KB LDS.
// All GEMMs use v_wmma_f32_16x16x32_bf16 (bf16 in, f32 accumulate), with
// 4-wide N register blocking so one LDS A-fragment feeds 4 WMMAs.
// ---------------------------------------------------------------------------

typedef __bf16 bf16;
typedef __attribute__((ext_vector_type(16))) __bf16 v16bf;
typedef __attribute__((ext_vector_type(8)))  float  v8f;

#define TT 64     // time steps
#define DM 512    // model dim
#define NNODES 2048
#define PT 520    // bf16 row pitch for [64][512] tiles (pad 8 -> conflict-free frag loads)
#define PV 72     // bf16 row pitch for transposed V [512][64]
#define PS 68     // f32 row pitch for scores [64][64]
#define PP 72     // bf16 row pitch for attn  [64][64]

union FragB16 { v16bf v; bf16 e[16]; uint4 q[2]; };

// Load a 16x32 bf16 fragment (A or B) whose lane-striped dimension is the LDS
// row and whose K dimension is contiguous within the row.
// ISA layout (7.12.2): lanes 0-15 -> row = lane, K = {k0..k0+7, k0+16..k0+23};
// lanes 16-31 -> row = lane-16, K shifted by 8.
__device__ __forceinline__ v16bf ld_frag(const bf16* rowblock, int pitch, int lane) {
    const int r  = lane & 15;
    const int ko = (lane >> 4) << 3;          // 0 or 8
    const bf16* p = rowblock + r * pitch + ko;
    FragB16 f;
    f.q[0] = *(const uint4*)(p);              // K = k0+ko .. +7
    f.q[1] = *(const uint4*)(p + 16);         // K = k0+16+ko .. +7
    return f.v;
}

__device__ __forceinline__ v8f wmma_bf16(v16bf a, v16bf b, v8f c) {
    return __builtin_amdgcn_wmma_f32_16x16x32_bf16(
        /*neg_a=*/false, a, /*neg_b=*/false, b,
        /*c_mod=*/(short)0, c, /*reuse_a=*/false, /*reuse_b=*/false);
}

// Store one 16x16 f32 accumulator tile as bf16 (C/D layout: VGPR r -> rows r / r+8).
__device__ __forceinline__ void st_tile_bf16(bf16* dst, int pitch, int lane, v8f acc) {
    const int col = lane & 15;
    const int rb  = (lane >> 4) << 3;
    bf16* d = dst + col;
#pragma unroll
    for (int r = 0; r < 8; ++r) d[(rb + r) * pitch] = (bf16)acc[r];
}

// ---------------------------------------------------------------------------
// Prep: convert weights to bf16 in workspace.
//   ws[0..3*D*D)  : Wq^T, Wk^T, Wv^T as [e][d]  (so B-fragments are K-contiguous)
//   ws[3*D*D ...) : lin_w as-is ([out,in] == [e][d] already)
// ---------------------------------------------------------------------------
__global__ void __launch_bounds__(256)
prep_weights_kernel(const float* __restrict__ Wq, const float* __restrict__ Wk,
                    const float* __restrict__ Wv, const float* __restrict__ Lw,
                    bf16* __restrict__ ws) {
    int i = blockIdx.x * blockDim.x + threadIdx.x;
    if (i >= DM * DM) return;
    int d = i / DM;           // source row    (K index of QKV GEMMs)
    int e = i % DM;           // source column (N index)
    ws[0 * DM * DM + e * DM + d] = (bf16)Wq[i];
    ws[1 * DM * DM + e * DM + d] = (bf16)Wk[i];
    ws[2 * DM * DM + e * DM + d] = (bf16)Wv[i];
    ws[3 * DM * DM + i]          = (bf16)Lw[i];
}

// ---------------------------------------------------------------------------
// Main kernel: one block (256 threads = 8 waves) per node.
// ---------------------------------------------------------------------------
__global__ void __launch_bounds__(256)
temporal_attn_kernel(const float* __restrict__ inputs,   // [T, N, D]
                     const float* __restrict__ pos_emb,  // [T, D]
                     const float* __restrict__ lin_b,    // [D]
                     const bf16*  __restrict__ wQt,      // [D][D] (= Wq^T)
                     const bf16*  __restrict__ wKt,
                     const bf16*  __restrict__ wVt,
                     const bf16*  __restrict__ wL,       // [D][D] (= lin_w)
                     float* __restrict__ out) {          // [N, T, D]
    __shared__ bf16  sT [TT * PT];   // temporal, bf16
    __shared__ bf16  sQ [TT * PT];   // Q, later reused as attention output O
    __shared__ bf16  sK [TT * PT];   // K
    __shared__ bf16  sVt[DM * PV];   // V transposed: [d][t]
    __shared__ float sS [TT * PS];   // scores f32
    __shared__ bf16  sP [TT * PP];   // softmax probs bf16

    const int n    = blockIdx.x;
    const int tid  = threadIdx.x;
    const int wave = tid >> 5;
    const int lane = tid & 31;

    // Warm L2 with weight streams (global_prefetch_b8).
    if (wave == 0) {
        __builtin_prefetch(wQt + lane * 8192, 0, 3);
        __builtin_prefetch(wKt + lane * 8192, 0, 3);
        __builtin_prefetch(wVt + lane * 8192, 0, 3);
        __builtin_prefetch(wL  + lane * 8192, 0, 3);
    }

    // ---- Phase 0: temporal = inputs[:, n, :] + pos_emb  ->  bf16 in LDS ----
    for (int idx4 = tid; idx4 < TT * DM / 4; idx4 += 256) {
        int t = idx4 >> 7;             // 128 float4 per row
        int c = (idx4 & 127) << 2;
        const float4 xi = *(const float4*)(inputs + ((size_t)t * NNODES + n) * DM + c);
        const float4 pe = *(const float4*)(pos_emb + t * DM + c);
        bf16* dst = sT + t * PT + c;
        dst[0] = (bf16)(xi.x + pe.x); dst[1] = (bf16)(xi.y + pe.y);
        dst[2] = (bf16)(xi.z + pe.z); dst[3] = (bf16)(xi.w + pe.w);
    }
    __syncthreads();

    // ---- Phase 1: Q/K/V = temporal x W   (V stored transposed) ----
    // Groups of 4 N-tiles: 3 mats x 4 M-tiles x 8 N-groups = 96 groups.
    for (int grp = wave; grp < 96; grp += 8) {
        const int mat = grp >> 5;          // 0..2
        const int rem = grp & 31;
        const int mt  = rem >> 3;          // 0..3
        const int ng  = rem & 7;           // N-group: nt = 4*ng .. 4*ng+3
        const bf16* W = (mat == 0) ? wQt : (mat == 1) ? wKt : wVt;
        v8f acc[4] = {};
        for (int kk = 0; kk < DM / 32; ++kk) {
            const v16bf a = ld_frag(sT + mt * 16 * PT + kk * 32, PT, lane);
#pragma unroll
            for (int j = 0; j < 4; ++j) {
                const v16bf b = ld_frag(W + (4 * ng + j) * 16 * DM + kk * 32, DM, lane);
                acc[j] = wmma_bf16(a, b, acc[j]);
            }
        }
        const int col = lane & 15;
        const int rb  = (lane >> 4) << 3;
        if (mat < 2) {
            bf16* base = (mat == 0 ? sQ : sK) + mt * 16 * PT;
#pragma unroll
            for (int j = 0; j < 4; ++j)
                st_tile_bf16(base + (4 * ng + j) * 16, PT, lane, acc[j]);
        } else {
            // transposed store: sVt[d][t]
#pragma unroll
            for (int j = 0; j < 4; ++j) {
                bf16* dst = sVt + ((4 * ng + j) * 16 + col) * PV + mt * 16 + rb;
#pragma unroll
                for (int r = 0; r < 8; ++r) dst[r] = (bf16)acc[j][r];
            }
        }
    }
    __syncthreads();

    // ---- Phase 2: scores = Q x K^T  (64x64, f32) ----
    // 4 M-tiles x 2 N-groups (2 tiles each) = 8 groups -> one per wave.
    for (int grp = wave; grp < 8; grp += 8) {
        const int mt = grp >> 1;
        const int ng = grp & 1;            // nt = 2*ng .. 2*ng+1
        v8f acc[2] = {};
        for (int kk = 0; kk < DM / 32; ++kk) {
            const v16bf a = ld_frag(sQ + mt * 16 * PT + kk * 32, PT, lane);
#pragma unroll
            for (int j = 0; j < 2; ++j) {
                const v16bf b = ld_frag(sK + (2 * ng + j) * 16 * PT + kk * 32, PT, lane);
                acc[j] = wmma_bf16(a, b, acc[j]);
            }
        }
        const int col = lane & 15;
        const int rb  = (lane >> 4) << 3;
#pragma unroll
        for (int j = 0; j < 2; ++j)
#pragma unroll
            for (int r = 0; r < 8; ++r)
                sS[(mt * 16 + rb + r) * PS + (2 * ng + j) * 16 + col] = acc[j][r];
    }
    __syncthreads();

    // ---- Phase 3: causal softmax over each row ----
    if (tid < TT) {
        const int t = tid;
        float m = -3.402823466e+38f;
        for (int k2 = 0; k2 <= t; ++k2) m = fmaxf(m, sS[t * PS + k2]);
        float sum = 0.f;
        for (int k2 = 0; k2 < TT; ++k2) {
            float ev = (k2 <= t) ? __expf(sS[t * PS + k2] - m) : 0.f;
            sum += ev;
            sS[t * PS + k2] = ev;
        }
        const float inv = 1.f / sum;
        for (int k2 = 0; k2 < TT; ++k2)
            sP[t * PP + k2] = (bf16)(sS[t * PS + k2] * inv);
    }
    __syncthreads();

    // ---- Phase 4: O = attn x V   (reuse sQ as O) ----
    // 4 M-tiles x 8 N-groups (4 tiles each) = 32 groups.
    bf16* sO = sQ;
    for (int grp = wave; grp < 32; grp += 8) {
        const int mt = grp >> 3;
        const int ng = grp & 7;
        v8f acc[4] = {};
        for (int kk = 0; kk < TT / 32; ++kk) {
            const v16bf a = ld_frag(sP + mt * 16 * PP + kk * 32, PP, lane);
#pragma unroll
            for (int j = 0; j < 4; ++j) {
                const v16bf b = ld_frag(sVt + (4 * ng + j) * 16 * PV + kk * 32, PV, lane);
                acc[j] = wmma_bf16(a, b, acc[j]);
            }
        }
#pragma unroll
        for (int j = 0; j < 4; ++j)
            st_tile_bf16(sO + mt * 16 * PT + (4 * ng + j) * 16, PT, lane, acc[j]);
    }
    __syncthreads();

    // ---- Phase 5: linear + ReLU + residuals, fused epilogue ----
    // final[t,e] = 0.5*(relu(O@lin_w^T + b) + O) + (inputs + pos_emb)
    for (int grp = wave; grp < 32; grp += 8) {
        const int mt = grp >> 3;
        const int ng = grp & 7;
        v8f acc[4] = {};
        for (int kk = 0; kk < DM / 32; ++kk) {
            const v16bf a = ld_frag(sO + mt * 16 * PT + kk * 32, PT, lane);
#pragma unroll
            for (int j = 0; j < 4; ++j) {
                const v16bf b = ld_frag(wL + (4 * ng + j) * 16 * DM + kk * 32, DM, lane);
                acc[j] = wmma_bf16(a, b, acc[j]);
            }
        }
        const int col = lane & 15;
        const int rb  = (lane >> 4) << 3;
#pragma unroll
        for (int j = 0; j < 4; ++j) {
            const int e = (4 * ng + j) * 16 + col;
            const float bias = lin_b[e];
#pragma unroll
            for (int r = 0; r < 8; ++r) {
                const int t = mt * 16 + rb + r;
                const float o  = (float)sO[t * PT + e];
                const float ff = fmaxf(acc[j][r] + bias, 0.f) + o;
                const float temporal =
                    inputs[((size_t)t * NNODES + n) * DM + e] + pos_emb[t * DM + e];
                out[((size_t)n * TT + t) * DM + e] = 0.5f * ff + temporal;
            }
        }
    }
}

// ---------------------------------------------------------------------------
extern "C" void kernel_launch(void* const* d_in, const int* in_sizes, int n_in,
                              void* d_out, int out_size, void* d_ws, size_t ws_size,
                              hipStream_t stream) {
    (void)in_sizes; (void)n_in; (void)out_size; (void)ws_size;
    const float* inputs  = (const float*)d_in[0];
    const float* pos_emb = (const float*)d_in[1];
    const float* Wq      = (const float*)d_in[2];
    const float* Wk      = (const float*)d_in[3];
    const float* Wv      = (const float*)d_in[4];
    const float* lin_w   = (const float*)d_in[5];
    const float* lin_b   = (const float*)d_in[6];

    bf16* ws  = (bf16*)d_ws;                  // needs 4*512*512*2 = 2 MB
    bf16* wQt = ws + 0 * DM * DM;
    bf16* wKt = ws + 1 * DM * DM;
    bf16* wVt = ws + 2 * DM * DM;
    bf16* wL  = ws + 3 * DM * DM;

    prep_weights_kernel<<<(DM * DM + 255) / 256, 256, 0, stream>>>(
        Wq, Wk, Wv, lin_w, ws);
    temporal_attn_kernel<<<NNODES, 256, 0, stream>>>(
        inputs, pos_emb, lin_b, wQt, wKt, wVt, wL, (float*)d_out);
}
